// Block_48052094108331
// MI455X (gfx1250) — matrix operand order repaired and checked by reading, not compiled, not run on hardware
//
#include <hip/hip_runtime.h>
#include <hip/hip_bf16.h>
#include <math.h>

typedef __attribute__((ext_vector_type(16))) __bf16 v16bf;
typedef __attribute__((ext_vector_type(8)))  float  v8f;
typedef __attribute__((ext_vector_type(4)))  unsigned int v4u;
typedef __attribute__((ext_vector_type(8)))  int v8i;
typedef __attribute__((ext_vector_type(4)))  int v4i;

union AFrag { unsigned short u[16]; v4u q[2]; v16bf v; };

// CDNA5 async / tensor data-mover availability (device pass only; safe fallback)
#if defined(__gfx1250__)
#if __has_builtin(__builtin_amdgcn_global_load_async_to_lds_b128)
#define HAVE_ASYNC_LDS 1
#endif
#if __has_builtin(__builtin_amdgcn_tensor_load_to_lds) && __has_builtin(__builtin_amdgcn_s_wait_tensorcnt)
#define HAVE_TDM 1
#endif
#endif

// async-to-LDS builtin takes typed v4i pointers: AS(1) global src, AS(3) LDS dst
typedef __attribute__((address_space(1))) v4i* g4_t;
typedef __attribute__((address_space(3))) v4i* l4_t;
__device__ __forceinline__ g4_t as_g4(const void* p) {
    return (g4_t)(unsigned long long)p;
}
__device__ __forceinline__ l4_t as_l4(void* p) {
    // generic LDS address: low 32 bits are the LDS offset on AMDGCN
    return (l4_t)(unsigned int)(unsigned long long)p;
}

__device__ __forceinline__ void wait_async() {
#if defined(HAVE_ASYNC_LDS)
#if __has_builtin(__builtin_amdgcn_s_wait_asynccnt)
    __builtin_amdgcn_s_wait_asynccnt(0);
#else
    asm volatile("s_wait_asynccnt 0x0" ::: "memory");
#endif
#endif
}

#if defined(HAVE_TDM)
// 1-D contiguous copy via the Tensor Data Mover.  D# per CDNA5 ISA ch.8:
// group0: count=1 | lds_addr | global_addr(57b) | type=2 ("image")
// group1: data_size=2B, tensor_dim0=tile_dim0=nelem, tensor_dim1=tile_dim1=1
__device__ __forceinline__ void tdm_load_1d(unsigned int lds_off,
                                            unsigned long long gaddr,
                                            unsigned int nelem) {
    v4u g0;
    g0.x = 1u;                                               // count=1, user D#
    g0.y = lds_off;                                          // lds_addr (bytes)
    g0.z = (unsigned int)(gaddr & 0xFFFFFFFFull);            // global_addr[31:0]
    g0.w = (unsigned int)((gaddr >> 32) & 0x1FFFFFFull) | 0x80000000u;  // [56:32] | type=2
    v8i g1;
    g1[0] = 0x00010000;                                      // data_size=1 (2 bytes)
    g1[1] = (int)((nelem & 0xFFFFu) << 16);                  // tensor_dim0[15:0]
    g1[2] = (int)(((nelem >> 16) & 0xFFFFu) | (1u << 16));   // tensor_dim0[31:16] | tensor_dim1=1
    g1[3] = (int)((nelem & 0xFFFFu) << 16);                  // tile_dim0 = nelem
    g1[4] = 1;                                               // tile_dim1 = 1
    g1[5] = (int)nelem;                                      // tensor_dim0_stride lo
    g1[6] = 0;
    g1[7] = 0;
    v4i gz4 = {0, 0, 0, 0};
    v8i gz8 = {0, 0, 0, 0, 0, 0, 0, 0};
    __builtin_amdgcn_tensor_load_to_lds(g0, g1, gz4, gz4, gz8, 0);
}
#endif

__device__ __forceinline__ unsigned short f2bf(float f) {
    unsigned int x = __float_as_uint(f);
    unsigned int r = (x + 0x7FFFu + ((x >> 16) & 1u)) >> 16;  // round-to-nearest-even
    return (unsigned short)r;
}

// ---------------------------------------------------------------------------
// LayerNorm (fp32 in) -> bf16 out.  One block (256 thr) per row of C=1024.
// ---------------------------------------------------------------------------
__global__ __launch_bounds__(256)
void layernorm_bf16(const float* __restrict__ x, const float* __restrict__ g,
                    const float* __restrict__ b, unsigned short* __restrict__ out) {
    const int row = blockIdx.x;
    const int t = threadIdx.x;
    const float4 v = reinterpret_cast<const float4*>(x + (size_t)row * 1024)[t];
    float s = v.x + v.y + v.z + v.w;
    float q = v.x * v.x + v.y * v.y + v.z * v.z + v.w * v.w;
#pragma unroll
    for (int m = 1; m < 32; m <<= 1) {
        s += __shfl_xor(s, m, 32);
        q += __shfl_xor(q, m, 32);
    }
    __shared__ float ss[8], sq[8];
    if ((t & 31) == 0) { ss[t >> 5] = s; sq[t >> 5] = q; }
    __syncthreads();
    float S = 0.f, Q = 0.f;
#pragma unroll
    for (int i = 0; i < 8; i++) { S += ss[i]; Q += sq[i]; }
    const float mu = S * (1.f / 1024.f);
    const float var = Q * (1.f / 1024.f) - mu * mu;
    const float rinv = rsqrtf(var + 1e-5f);
    const float4 gg = reinterpret_cast<const float4*>(g)[t];
    const float4 bb = reinterpret_cast<const float4*>(b)[t];
    unsigned short* o = out + (size_t)row * 1024 + t * 4;
    o[0] = f2bf((v.x - mu) * rinv * gg.x + bb.x);
    o[1] = f2bf((v.y - mu) * rinv * gg.y + bb.y);
    o[2] = f2bf((v.z - mu) * rinv * gg.z + bb.z);
    o[3] = f2bf((v.w - mu) * rinv * gg.w + bb.w);
}

// ---------------------------------------------------------------------------
// bf16 WMMA GEMM:  C[M,N] = A(bf16)[M,K] @ W(f32->bf16)[K,N] + bias
// Block tile 128x128x32, 8 waves each computing 64x32 (4x2 wmma tiles).
// A tile staged via async-to-LDS (copy overlaps the B convert); B tile stored
// col-major in LDS so both fragment gathers are contiguous 16B ds loads.
// OUT_MODE: 0 = f32 + residual add   1 = bf16 (+ReLU)   2 = QKV scatter bf16
// ---------------------------------------------------------------------------
template<int OUT_MODE, bool RELU>
__global__ __launch_bounds__(256)
void gemm_wmma(const unsigned short* __restrict__ A,
               const float* __restrict__ Bw,
               const float* __restrict__ bias,
               const float* __restrict__ resid,
               void* __restrict__ outp,
               int M, int Nn, int K) {
    __shared__ unsigned short As[128 * 32];   // [row][k]
    __shared__ unsigned short Bs[128 * 32];   // [col][k]  (col-major tile)
    const int t = threadIdx.x;
    const int lane = t & 31;
    const int w = t >> 5;
    const int wm = (w & 1) * 64;
    const int wn = (w >> 1) * 32;
    const int m0 = blockIdx.y * 128;
    const int n0 = blockIdx.x * 128;
    const int l15 = lane & 15;
    const int hi8 = (lane >= 16) ? 8 : 0;
    const int hi16 = (lane >= 16) ? 16 : 0;

    v8f acc[4][2];
    const v8f zero8 = {0.f, 0.f, 0.f, 0.f, 0.f, 0.f, 0.f, 0.f};
#pragma unroll
    for (int mi = 0; mi < 4; mi++)
#pragma unroll
        for (int ni = 0; ni < 2; ni++) acc[mi][ni] = zero8;

    for (int k0 = 0; k0 < K; k0 += 32) {
        {   // stage A tile: 128x32 bf16 (async when available)
            const int idx = t * 16;
            const int row = idx >> 5, col = idx & 31;
            const unsigned short* ag = A + (size_t)(m0 + row) * K + k0 + col;
#if defined(HAVE_ASYNC_LDS)
            __builtin_amdgcn_global_load_async_to_lds_b128(as_g4(ag), as_l4(&As[idx]), 0, 0);
            __builtin_amdgcn_global_load_async_to_lds_b128(as_g4(ag + 8), as_l4(&As[idx + 8]), 0, 0);
#else
            const uint4* src = reinterpret_cast<const uint4*>(ag);
            uint4* dst = reinterpret_cast<uint4*>(&As[idx]);
            dst[0] = src[0];
            dst[1] = src[1];
#endif
            if (k0 + 32 < K) __builtin_prefetch(ag + 32, 0, 3);
        }
        {   // stage B tile: 32x128 f32 -> bf16, transposed into [col][k]
            const int idx = t * 16;
            const int r = idx >> 7, c = idx & 127;
            const float* bg = Bw + (size_t)(k0 + r) * Nn + n0 + c;
            const float4* src = reinterpret_cast<const float4*>(bg);
#pragma unroll
            for (int j = 0; j < 4; j++) {
                float4 f = src[j];
                Bs[(c + j * 4 + 0) * 32 + r] = f2bf(f.x);
                Bs[(c + j * 4 + 1) * 32 + r] = f2bf(f.y);
                Bs[(c + j * 4 + 2) * 32 + r] = f2bf(f.z);
                Bs[(c + j * 4 + 3) * 32 + r] = f2bf(f.w);
            }
            if (k0 + 32 < K) __builtin_prefetch(bg + (size_t)32 * Nn, 0, 3);
        }
        wait_async();
        __syncthreads();

        AFrag af[4], bf[2];
#pragma unroll
        for (int mi = 0; mi < 4; mi++) {
            const int row = wm + mi * 16 + l15;
            af[mi].q[0] = *reinterpret_cast<const v4u*>(&As[row * 32 + hi8]);
            af[mi].q[1] = *reinterpret_cast<const v4u*>(&As[row * 32 + 16 + hi8]);
        }
#pragma unroll
        for (int ni = 0; ni < 2; ni++) {
            const int col = wn + ni * 16 + l15;
            bf[ni].q[0] = *reinterpret_cast<const v4u*>(&Bs[col * 32 + hi8]);
            bf[ni].q[1] = *reinterpret_cast<const v4u*>(&Bs[col * 32 + 16 + hi8]);
        }
#pragma unroll
        for (int mi = 0; mi < 4; mi++)
#pragma unroll
            for (int ni = 0; ni < 2; ni++)
                acc[mi][ni] = __builtin_amdgcn_wmma_f32_16x16x32_bf16(
                    false, af[mi].v, false, bf[ni].v, (short)0, acc[mi][ni], false, false);
        __syncthreads();
    }

#pragma unroll
    for (int mi = 0; mi < 4; mi++)
#pragma unroll
        for (int ni = 0; ni < 2; ni++)
#pragma unroll
            for (int v = 0; v < 8; v++) {
                const int gr = m0 + wm + mi * 16 + v + hi8;
                const int gc = n0 + wn + ni * 16 + l15;
                float val = acc[mi][ni][v] + bias[gc];
                if (RELU) val = fmaxf(val, 0.f);
                if (OUT_MODE == 0) {
                    ((float*)outp)[(size_t)gr * Nn + gc] = val + resid[(size_t)gr * Nn + gc];
                } else if (OUT_MODE == 1) {
                    ((unsigned short*)outp)[(size_t)gr * Nn + gc] = f2bf(val);
                } else {
                    const int which = gc >> 10, rem = gc & 1023;
                    const int head = rem >> 6, d = rem & 63;
                    const int b = gr >> 11, n = gr & 2047;
                    const size_t dst = (size_t)which * 4194304 +
                                       (size_t)((b * 16 + head) * 2048 + n) * 64 + d;
                    ((unsigned short*)outp)[dst] = f2bf(val);
                }
            }
}

// ---------------------------------------------------------------------------
// Flash attention: grid (N/128, B*H); 8 waves x 16 query rows per block.
// K/V tiles (contiguous 8KB each) staged via the Tensor Data Mover when
// available (wave 0 issues 1-D descriptors, s_wait_tensorcnt, barrier),
// else async-to-LDS, else plain loads.  WMMA for QK^T and PV; online softmax.
// ---------------------------------------------------------------------------
__global__ __launch_bounds__(256)
void flash_attn(const unsigned short* __restrict__ Q,
                const unsigned short* __restrict__ Kb,
                const unsigned short* __restrict__ Vb,
                const float* __restrict__ x,
                float* __restrict__ xattn) {
    __shared__ unsigned short Ks[64 * 64];
    __shared__ unsigned short Vs[64 * 64];
    __shared__ unsigned short Ps[8][16 * 64];
    const int t = threadIdx.x, lane = t & 31, w = t >> 5, l15 = lane & 15;
    const int hi8 = (lane >= 16) ? 8 : 0, hi16 = (lane >= 16) ? 16 : 0;
    const int bh = blockIdx.y;
    const int q0 = blockIdx.x * 128 + w * 16;
    const size_t base = (size_t)bh * 2048 * 64;
    const float SCALE = 0.03125f;  // 1/sqrt(C) = 1/32 (faithful to source)

    AFrag qf[2];
    {
        const unsigned short* qrow = Q + base + (size_t)(q0 + l15) * 64;
#pragma unroll
        for (int kf = 0; kf < 2; kf++) {
            qf[kf].q[0] = *reinterpret_cast<const v4u*>(qrow + hi8 + kf * 32);
            qf[kf].q[1] = *reinterpret_cast<const v4u*>(qrow + 16 + hi8 + kf * 32);
        }
    }

    float mst[8], lst[8];
    v8f oacc[4];
    const v8f zero8 = {0.f, 0.f, 0.f, 0.f, 0.f, 0.f, 0.f, 0.f};
#pragma unroll
    for (int v = 0; v < 8; v++) { mst[v] = -1e30f; lst[v] = 0.f; }
#pragma unroll
    for (int tI = 0; tI < 4; tI++) oacc[tI] = zero8;

    for (int kt = 0; kt < 2048; kt += 64) {
#if defined(HAVE_TDM)
        if (w == 0) {
            tdm_load_1d((unsigned int)(unsigned long long)&Ks[0],
                        (unsigned long long)(Kb + base + (size_t)kt * 64), 4096);
            tdm_load_1d((unsigned int)(unsigned long long)&Vs[0],
                        (unsigned long long)(Vb + base + (size_t)kt * 64), 4096);
            __builtin_amdgcn_s_wait_tensorcnt(0);
        }
        __syncthreads();
#else
        {
            const int idx = t * 16;
            const int row = idx >> 6, col = idx & 63;
            const unsigned short* kg = Kb + base + (size_t)(kt + row) * 64 + col;
            const unsigned short* vg = Vb + base + (size_t)(kt + row) * 64 + col;
#if defined(HAVE_ASYNC_LDS)
            __builtin_amdgcn_global_load_async_to_lds_b128(as_g4(kg), as_l4(&Ks[idx]), 0, 0);
            __builtin_amdgcn_global_load_async_to_lds_b128(as_g4(kg + 8), as_l4(&Ks[idx + 8]), 0, 0);
            __builtin_amdgcn_global_load_async_to_lds_b128(as_g4(vg), as_l4(&Vs[idx]), 0, 0);
            __builtin_amdgcn_global_load_async_to_lds_b128(as_g4(vg + 8), as_l4(&Vs[idx + 8]), 0, 0);
#else
            const uint4* ks = reinterpret_cast<const uint4*>(kg);
            uint4* kd = reinterpret_cast<uint4*>(&Ks[idx]);
            kd[0] = ks[0]; kd[1] = ks[1];
            const uint4* vs = reinterpret_cast<const uint4*>(vg);
            uint4* vd = reinterpret_cast<uint4*>(&Vs[idx]);
            vd[0] = vs[0]; vd[1] = vs[1];
#endif
        }
        wait_async();
        __syncthreads();
#endif

        // scores: 16 rows x 64 keys
        v8f sacc[4];
#pragma unroll
        for (int ni = 0; ni < 4; ni++) sacc[ni] = zero8;
#pragma unroll
        for (int ni = 0; ni < 4; ni++) {
            const int n = ni * 16 + l15;
#pragma unroll
            for (int kf = 0; kf < 2; kf++) {
                AFrag bfr;
                const unsigned short* kp = &Ks[n * 64 + hi8 + kf * 32];
                bfr.q[0] = *reinterpret_cast<const v4u*>(kp);
                bfr.q[1] = *reinterpret_cast<const v4u*>(kp + 16);
                sacc[ni] = __builtin_amdgcn_wmma_f32_16x16x32_bf16(
                    false, qf[kf].v, false, bfr.v, (short)0, sacc[ni], false, false);
            }
        }

        // online softmax (row stats per VGPR slot, reduced over 16-lane groups)
        float alpha[8];
#pragma unroll
        for (int v = 0; v < 8; v++) {
            float mx = -1e30f;
#pragma unroll
            for (int ni = 0; ni < 4; ni++) {
                const float sv = sacc[ni][v] * SCALE;
                sacc[ni][v] = sv;
                mx = fmaxf(mx, sv);
            }
#pragma unroll
            for (int m = 1; m < 16; m <<= 1) mx = fmaxf(mx, __shfl_xor(mx, m, 32));
            const float mnew = fmaxf(mst[v], mx);
            alpha[v] = __expf(mst[v] - mnew);
            float rs = 0.f;
#pragma unroll
            for (int ni = 0; ni < 4; ni++) {
                const float p = __expf(sacc[ni][v] - mnew);
                sacc[ni][v] = p;
                rs += p;
            }
#pragma unroll
            for (int m = 1; m < 16; m <<= 1) rs += __shfl_xor(rs, m, 32);
            lst[v] = lst[v] * alpha[v] + rs;
            mst[v] = mnew;
        }

        // re-layout P (C-layout -> A-layout) via per-wave LDS scratch
#pragma unroll
        for (int ni = 0; ni < 4; ni++)
#pragma unroll
            for (int v = 0; v < 8; v++)
                Ps[w][(v + hi8) * 64 + ni * 16 + l15] = f2bf(sacc[ni][v]);
        __syncthreads();

#pragma unroll
        for (int tI = 0; tI < 4; tI++)
#pragma unroll
            for (int v = 0; v < 8; v++) oacc[tI][v] *= alpha[v];

        // P @ V
#pragma unroll
        for (int kf = 0; kf < 2; kf++) {
            AFrag pf;
            pf.q[0] = *reinterpret_cast<const v4u*>(&Ps[w][l15 * 64 + hi8 + kf * 32]);
            pf.q[1] = *reinterpret_cast<const v4u*>(&Ps[w][l15 * 64 + 16 + hi8 + kf * 32]);
#pragma unroll
            for (int tI = 0; tI < 4; tI++) {
                AFrag bfr;
                const int col = tI * 16 + l15;
#pragma unroll
                for (int e = 0; e < 16; e++) bfr.u[e] = Vs[((e & 7) + ((e >> 3) << 4) + hi8 + kf * 32) * 64 + col];
                oacc[tI] = __builtin_amdgcn_wmma_f32_16x16x32_bf16(
                    false, pf.v, false, bfr.v, (short)0, oacc[tI], false, false);
            }
        }
        __syncthreads();
    }

    // epilogue: reference reshape [B,H,N,hd] -> [B,N,C] without head transpose:
    // row' = b*2048 + h*128 + n/16 ; col' = (n%16)*64 + d.  Fuse x + attn.
    const int b = bh >> 4, h = bh & 15;
#pragma unroll
    for (int tI = 0; tI < 4; tI++)
#pragma unroll
        for (int v = 0; v < 8; v++) {
            const int n = q0 + v + hi8;
            const int d = tI * 16 + l15;
            const float o = oacc[tI][v] / lst[v];
            const int rowp = b * 2048 + h * 128 + (n >> 4);
            const int colp = ((n & 15) << 6) + d;
            const size_t ix = (size_t)rowp * 1024 + colp;
            xattn[ix] = x[ix] + o;
        }
}

// ---------------------------------------------------------------------------
extern "C" void kernel_launch(void* const* d_in, const int* in_sizes, int n_in,
                              void* d_out, int out_size, void* d_ws, size_t ws_size,
                              hipStream_t stream) {
    const float* x     = (const float*)d_in[0];
    const float* qkv_w = (const float*)d_in[1];
    const float* qkv_b = (const float*)d_in[2];
    const float* w1    = (const float*)d_in[3];
    const float* b1    = (const float*)d_in[4];
    const float* w2    = (const float*)d_in[5];
    const float* b2    = (const float*)d_in[6];
    const float* ln1_g = (const float*)d_in[7];
    const float* ln1_b = (const float*)d_in[8];
    const float* ln2_g = (const float*)d_in[9];
    const float* ln2_b = (const float*)d_in[10];

    char* ws = (char*)d_ws;
    unsigned short* qkv  = (unsigned short*)ws;                         // 24 MB: [3][B][H][N][64] bf16
    unsigned short* hbuf = (unsigned short*)(ws + 25165824);            //  8 MB: LN output bf16
    float*          xatt = (float*)(ws + 25165824 + 8388608);           // 16 MB: x + attn (fp32)
    unsigned short* ffb  = (unsigned short*)(ws + 25165824 + 8388608 + 16777216); // 32 MB: FF hidden

    layernorm_bf16<<<4096, 256, 0, stream>>>(x, ln1_g, ln1_b, hbuf);
    gemm_wmma<2, false><<<dim3(24, 32), 256, 0, stream>>>(hbuf, qkv_w, qkv_b, nullptr, qkv,
                                                          4096, 3072, 1024);
    flash_attn<<<dim3(16, 32), 256, 0, stream>>>(qkv, qkv + 4194304, qkv + 8388608, x, xatt);
    layernorm_bf16<<<4096, 256, 0, stream>>>(xatt, ln2_g, ln2_b, hbuf);
    gemm_wmma<1, true><<<dim3(32, 32), 256, 0, stream>>>(hbuf, w1, b1, nullptr, ffb,
                                                         4096, 4096, 1024);
    gemm_wmma<0, false><<<dim3(8, 32), 256, 0, stream>>>(ffb, w2, b2, xatt, d_out,
                                                         4096, 1024, 4096);
}
